// Block_47261820125599
// MI455X (gfx1250) — matrix operand order repaired and checked by reading, not compiled, not run on hardware
//
#include <hip/hip_runtime.h>
#include <math.h>

// ---------------------------------------------------------------------------
// Graph-transformer block for MI455X (gfx1250, wave32, WMMA 16x16x32 f16,
// TDM-staged weights in LDS).
// ---------------------------------------------------------------------------

#define NN 4096      // nodes
#define DM 256       // model dim
#define NH 8         // heads
#define DH 32        // head dim

typedef __attribute__((ext_vector_type(16))) _Float16 v16h;
typedef __attribute__((ext_vector_type(8)))  float    v8f;
typedef __attribute__((ext_vector_type(4)))  int      v4i;
typedef __attribute__((ext_vector_type(8)))  int      v8i;

union AF { v16h v; uint4 q[2]; };

__device__ inline v8f zerov() { v8f z; for (int i = 0; i < 8; ++i) z[i] = 0.f; return z; }

__device__ inline v8f wmma_f16(v16h a, v16h b, v8f c) {
  return __builtin_amdgcn_wmma_f32_16x16x32_f16(false, a, false, b, (short)0, c, false, false);
}

// A operand, 16x32 f16, rows r0..r0+15, k0..k0+31, per ISA layout:
// lane<16: row=lane, halves = K[k0..k0+7], K[k0+16..k0+23]
// lane>=16: row=lane-16, halves = K[k0+8..k0+15], K[k0+24..k0+31]
__device__ inline v16h load_a_frag(const _Float16* A, int lda, int r0, int k0, int lane) {
  int row = r0 + (lane & 15);
  int kb  = k0 + ((lane >> 4) << 3);
  const _Float16* p = A + (size_t)row * lda + kb;
  AF f;
  f.q[0] = *(const uint4*)(p);
  f.q[1] = *(const uint4*)(p + 16);
  return f.v;
}

// B operand, 32x16 f16, K rows k0..k0+31, cols c0..c0+15:
// lane l holds N=c0+l%16, halves = K[k0+8h..+7], K[k0+16+8h..+7] (h=l/16)
__device__ inline v16h load_b_frag(const _Float16* B, int ldb, int k0, int c0, int lane) {
  int n  = c0 + (lane & 15);
  int kb = k0 + ((lane >> 4) << 3);
  v16h f;
  for (int s = 0; s < 8; ++s) {
    f[s]     = B[(size_t)(kb + s) * ldb + n];
    f[s + 8] = B[(size_t)(kb + 16 + s) * ldb + n];
  }
  return f;
}

__device__ inline float hmax16(float v) {
  for (int m = 1; m < 16; m <<= 1) v = fmaxf(v, __shfl_xor(v, m, 32));
  return v;
}
__device__ inline float hsum16(float v) {
  for (int m = 1; m < 16; m <<= 1) v += __shfl_xor(v, m, 32);
  return v;
}
__device__ inline float wsum32(float v) {
  for (int m = 1; m < 32; m <<= 1) v += __shfl_xor(v, m, 32);
  return v;
}

// ---------------------------------------------------------------------------
// Tensor Data Mover: DMA a K x Nc f16 row-major matrix from global into LDS.
// D# built per CDNA5 ISA: 2D tensor, data_size=8B,
// groups 2/3 NULL (<=2D), count=1, type=2. Issue once per workgroup.
// ---------------------------------------------------------------------------
__device__ inline void tdm_load_matrix_to_lds(const _Float16* B, unsigned lds_off,
                                              int K, int Nc) {
  unsigned long long gaddr = (unsigned long long)B;
  int rowu = (Nc * 2) / 8;  // row length in 8-byte units
  v4i g0;
  g0[0] = 1;                                           // count=1 (valid), user mode
  g0[1] = (int)lds_off;                                // lds_addr [63:32]
  g0[2] = (int)(unsigned)(gaddr & 0xffffffffull);      // global_addr lo
  g0[3] = (int)((unsigned)((gaddr >> 32) & 0x01ffffffull) | 0x80000000u); // hi + type=2
  v8i g1;
  g1[0] = 3 << 16;          // wg_mask=0, data_size=3 (8B), no pad/iterate/barrier
  g1[1] = rowu << 16;       // tensor_dim0[15:0] @ bits 63:48
  g1[2] = K << 16;          // tensor_dim1[15:0] @ bits 95:80
  g1[3] = rowu << 16;       // tile_dim0 @ bits 127:112
  g1[4] = K;                // tile_dim1 @ bits 143:128, tile_dim2=0
  g1[5] = rowu;             // tensor_dim0_stride[31:0] @ bits 191:160
  g1[6] = 0;                // stride hi + tensor_dim1_stride lo
  g1[7] = 0;
  asm volatile("tensor_load_to_lds %0, %1" :: "s"(g0), "s"(g1) : "memory");
  __builtin_amdgcn_s_wait_tensorcnt(0);
}

// ---------------------------------------------------------------------------
// Small utility kernels
// ---------------------------------------------------------------------------
__global__ void zero_u32(unsigned int* p, int n) {
  for (int i = blockIdx.x * blockDim.x + threadIdx.x; i < n; i += gridDim.x * blockDim.x)
    p[i] = 0u;
}

__global__ void build_mask(const long long* __restrict__ e, int E,
                           unsigned int* __restrict__ words) {
  int i = blockIdx.x * blockDim.x + threadIdx.x;
  if (i >= E) return;
  int row = (int)e[i];
  int col = (int)e[(size_t)E + i];
  size_t bidx = (size_t)row * NN + col;
  atomicOr(&words[bidx >> 5], 1u << (bidx & 31));
}

__global__ void cvt_f16(const float* __restrict__ in, _Float16* __restrict__ out, int n) {
  int i = blockIdx.x * blockDim.x + threadIdx.x;
  if (i < n) out[i] = (_Float16)in[i];
}

// kT[h][dh][n] = k[n][h*32+dh]  (so attention B-fragments are contiguous)
__global__ void transpose_k(const _Float16* __restrict__ k, _Float16* __restrict__ kT) {
  int idx = blockIdx.x * blockDim.x + threadIdx.x;  // over NN*DM
  int n = idx >> 8;
  int d = idx & 255;
  kT[(size_t)d * NN + n] = k[idx];
}

// ---------------------------------------------------------------------------
// LayerNorm: one wave per row (256 elems -> 8 per lane), f16 output
// ---------------------------------------------------------------------------
struct __attribute__((aligned(16))) H8 { _Float16 h[8]; };

__global__ __launch_bounds__(256) void ln_kernel(const float* __restrict__ x,
                                                 const float* __restrict__ g,
                                                 const float* __restrict__ b,
                                                 _Float16* __restrict__ out) {
  int lane = threadIdx.x & 31;
  int wave = threadIdx.x >> 5;
  int row  = blockIdx.x * 8 + wave;
  const float* xr = x + (size_t)row * DM;
  float4 a0 = *(const float4*)(xr + lane * 8);
  float4 a1 = *(const float4*)(xr + lane * 8 + 4);
  float xs[8] = {a0.x, a0.y, a0.z, a0.w, a1.x, a1.y, a1.z, a1.w};
  float s = 0.f;
  for (int j = 0; j < 8; ++j) s += xs[j];
  s = wsum32(s);
  float mu = s * (1.f / DM);
  float vs = 0.f;
  for (int j = 0; j < 8; ++j) { float d = xs[j] - mu; vs += d * d; }
  vs = wsum32(vs);
  float rs = rsqrtf(vs * (1.f / DM) + 1e-5f);
  H8 o;
  int c0 = lane * 8;
  for (int j = 0; j < 8; ++j)
    o.h[j] = (_Float16)((xs[j] - mu) * rs * g[c0 + j] + b[c0 + j]);
  *(H8*)(out + (size_t)row * DM + c0) = o;
}

// ---------------------------------------------------------------------------
// Generic WMMA GEMM: C[M x Nc] = A[M x K] * B[K x Nc] + bias, optional exact
// GELU, optional fp32 residual, f16 and/or f32 outputs. One wave = 32x32 tile.
// B (the weight matrix, 128-256KB) is TDM-staged into LDS once per workgroup
// and B fragments are gathered from LDS one f16 element at a time (stride Nc).
// ---------------------------------------------------------------------------
__device__ inline float gelu_exact(float x) {
  return 0.5f * x * (1.f + erff(x * 0.70710678118654752f));
}

__global__ __launch_bounds__(256) void gemm_kernel(
    const _Float16* __restrict__ A, const _Float16* __restrict__ B,
    const float* __restrict__ bias, const float* __restrict__ resid,
    _Float16* __restrict__ out16, float* __restrict__ out32,
    int K, int Nc, int act) {
  extern __shared__ _Float16 ldsB[];  // K x Nc
  int lane = threadIdx.x & 31;
  int wave = threadIdx.x >> 5;
  int halfw = (lane >> 4) & 1;
  int tiles_n = Nc >> 5;
  int tile = blockIdx.x * 8 + wave;
  int tm = tile / tiles_n, tn = tile % tiles_n;
  int r0 = tm * 32, c0 = tn * 32;

  // Stage entire B into LDS via the Tensor Data Mover (one issue per block).
  if (wave == 0) {
    unsigned lds_off = (unsigned)(unsigned long long)(const void*)ldsB;
    tdm_load_matrix_to_lds(B, lds_off, K, Nc);
  }
  __syncthreads();

  v8f c00 = zerov(), c01 = zerov(), c10 = zerov(), c11 = zerov();
  for (int k0 = 0; k0 < K; k0 += 32) {
    v16h a0 = load_a_frag(A, K, r0, k0, lane);
    v16h a1 = load_a_frag(A, K, r0 + 16, k0, lane);
    v16h b0 = load_b_frag(ldsB, Nc, k0, c0, lane);
    v16h b1 = load_b_frag(ldsB, Nc, k0, c0 + 16, lane);
    c00 = wmma_f16(a0, b0, c00);
    c01 = wmma_f16(a0, b1, c01);
    c10 = wmma_f16(a1, b0, c10);
    c11 = wmma_f16(a1, b1, c11);
  }
  int col0 = c0 + (lane & 15);
  int col1 = col0 + 16;
  float bs0 = bias[col0], bs1 = bias[col1];
  for (int a = 0; a < 2; ++a) {
    v8f* cr0 = a ? &c10 : &c00;
    v8f* cr1 = a ? &c11 : &c01;
    for (int i = 0; i < 8; ++i) {
      int row = r0 + a * 16 + i + 8 * halfw;
      float v0 = (*cr0)[i] + bs0;
      float v1 = (*cr1)[i] + bs1;
      if (act) { v0 = gelu_exact(v0); v1 = gelu_exact(v1); }
      if (resid) {
        v0 += resid[(size_t)row * Nc + col0];
        v1 += resid[(size_t)row * Nc + col1];
      }
      if (out16) {
        out16[(size_t)row * Nc + col0] = (_Float16)v0;
        out16[(size_t)row * Nc + col1] = (_Float16)v1;
      }
      if (out32) {
        out32[(size_t)row * Nc + col0] = v0;
        out32[(size_t)row * Nc + col1] = v1;
      }
    }
  }
}

// ---------------------------------------------------------------------------
// Masked attention, flash-style. One wave per (head, 16-row block).
// Scores for a 32-key chunk = 2 WMMAs (DH=32 == native K), PV = 2 WMMAs.
// ---------------------------------------------------------------------------
__global__ __launch_bounds__(32) void attn_kernel(
    const _Float16* __restrict__ q, const _Float16* __restrict__ kT,
    const _Float16* __restrict__ v, const unsigned char* __restrict__ maskb,
    _Float16* __restrict__ y) {
  int lane = threadIdx.x & 31;
  int halfw = (lane >> 4) & 1;
  int r0 = blockIdx.x * 16;
  int hh = blockIdx.y;
  const float scale = 0.17677669529663687f;  // 1/sqrt(32)
  const float NEGINF = -__builtin_inff();

  __shared__ __align__(16) _Float16 pl[16 * 32];

  v16h qf = load_a_frag(q, DM, r0, hh * DH, lane);
  const _Float16* kTh = kT + (size_t)hh * DH * NN;

  v8f o0 = zerov(), o1 = zerov();
  float mrun[8], lrun[8];
  for (int i = 0; i < 8; ++i) { mrun[i] = NEGINF; lrun[i] = 0.f; }

  for (int key0 = 0; key0 < NN; key0 += 32) {
    // prefetch next chunk of the mask bitmap (global_prefetch_b8)
    __builtin_prefetch(maskb + (((size_t)(r0 + (lane & 15)) * NN + key0 + 32) >> 3), 0, 1);

    v16h kt0 = load_b_frag(kTh, NN, 0, key0, lane);
    v16h kt1 = load_b_frag(kTh, NN, 0, key0 + 16, lane);
    v8f s0 = wmma_f16(qf, kt0, zerov());
    v8f s1 = wmma_f16(qf, kt1, zerov());

    // scale + mask (bit set == edge == attended; else -inf)
    for (int t = 0; t < 2; ++t) {
      v8f* sp = t ? &s1 : &s0;
      int col = key0 + t * 16 + (lane & 15);
      for (int i = 0; i < 8; ++i) {
        int row = r0 + i + 8 * halfw;
        size_t bidx = (size_t)row * NN + col;
        float sv = (*sp)[i] * scale;
        unsigned char mb = maskb[bidx >> 3];
        if (!((mb >> (bidx & 7)) & 1)) sv = NEGINF;
        (*sp)[i] = sv;
      }
    }

    // online softmax per row (row constant across a half-wave for fixed i)
    for (int i = 0; i < 8; ++i) {
      float m = hmax16(fmaxf(s0[i], s1[i]));
      float mnew = fmaxf(mrun[i], m);
      float corr = (mrun[i] == NEGINF) ? 0.f : __expf(mrun[i] - mnew);
      float p0 = (mnew == NEGINF) ? 0.f : __expf(s0[i] - mnew);
      float p1 = (mnew == NEGINF) ? 0.f : __expf(s1[i] - mnew);
      float ps = hsum16(p0 + p1);
      lrun[i] = lrun[i] * corr + ps;
      mrun[i] = mnew;
      o0[i] *= corr;
      o1[i] *= corr;
      s0[i] = p0;
      s1[i] = p1;
    }

    // P: C-layout -> A-layout via LDS transpose
    __syncthreads();
    for (int t = 0; t < 2; ++t) {
      int col = t * 16 + (lane & 15);
      for (int i = 0; i < 8; ++i)
        pl[(i + 8 * halfw) * 32 + col] = (_Float16)(t ? s1[i] : s0[i]);
    }
    __syncthreads();
    v16h pf = load_a_frag(pl, 32, 0, 0, lane);

    v16h vf0 = load_b_frag(v, DM, key0, hh * DH, lane);
    v16h vf1 = load_b_frag(v, DM, key0, hh * DH + 16, lane);
    o0 = wmma_f16(pf, vf0, o0);
    o1 = wmma_f16(pf, vf1, o1);
  }

  for (int i = 0; i < 8; ++i) {
    float inv = lrun[i] > 0.f ? 1.f / lrun[i] : 0.f;
    int row = r0 + i + 8 * halfw;
    int c = hh * DH + (lane & 15);
    y[(size_t)row * DM + c]      = (_Float16)(o0[i] * inv);
    y[(size_t)row * DM + c + 16] = (_Float16)(o1[i] * inv);
  }
}

// ---------------------------------------------------------------------------
// Orchestration
// ---------------------------------------------------------------------------
extern "C" void kernel_launch(void* const* d_in, const int* in_sizes, int n_in,
                              void* d_out, int out_size, void* d_ws, size_t ws_size,
                              hipStream_t stream) {
  const float*     x   = (const float*)d_in[0];
  const long long* ei  = (const long long*)d_in[1];
  const float* Wq = (const float*)d_in[2];  const float* bq = (const float*)d_in[3];
  const float* Wk = (const float*)d_in[4];  const float* bk = (const float*)d_in[5];
  const float* Wv = (const float*)d_in[6];  const float* bv = (const float*)d_in[7];
  const float* Wo = (const float*)d_in[8];  const float* bo = (const float*)d_in[9];
  const float* g1 = (const float*)d_in[10]; const float* b1 = (const float*)d_in[11];
  const float* g2 = (const float*)d_in[12]; const float* b2 = (const float*)d_in[13];
  const float* Wm1 = (const float*)d_in[14]; const float* bm1 = (const float*)d_in[15];
  const float* Wm2 = (const float*)d_in[16]; const float* bm2 = (const float*)d_in[17];

  char* ws = (char*)d_ws;
  size_t off = 0;
  auto alloc = [&](size_t bytes) {
    size_t o = off;
    off = (off + bytes + 255) & ~(size_t)255;
    return o;
  };
  size_t mask_off = alloc((size_t)NN * NN / 8);        // 2 MB bitmap
  size_t h_off    = alloc((size_t)NN * DM * 2);        // LN1 out f16
  size_t q_off    = alloc((size_t)NN * DM * 2);
  size_t k_off    = alloc((size_t)NN * DM * 2);
  size_t v_off    = alloc((size_t)NN * DM * 2);
  size_t kt_off   = alloc((size_t)NN * DM * 2);        // K^T per head
  size_t y_off    = alloc((size_t)NN * DM * 2);        // attention out f16
  size_t x1_off   = alloc((size_t)NN * DM * 4);        // x + y@Wo (fp32)
  size_t h2_off   = alloc((size_t)NN * DM * 2);        // LN2 out f16
  size_t m_off    = alloc((size_t)NN * 2 * DM * 2);    // GELU(h2@Wm1) f16
  size_t wq_off   = alloc((size_t)DM * DM * 2);
  size_t wk_off   = alloc((size_t)DM * DM * 2);
  size_t wv_off   = alloc((size_t)DM * DM * 2);
  size_t wo_off   = alloc((size_t)DM * DM * 2);
  size_t wm1_off  = alloc((size_t)DM * 2 * DM * 2);
  size_t wm2_off  = alloc((size_t)2 * DM * DM * 2);

  unsigned int* maskw = (unsigned int*)(ws + mask_off);
  const unsigned char* maskb = (const unsigned char*)(ws + mask_off);
  _Float16* hf   = (_Float16*)(ws + h_off);
  _Float16* qf   = (_Float16*)(ws + q_off);
  _Float16* kf   = (_Float16*)(ws + k_off);
  _Float16* vf   = (_Float16*)(ws + v_off);
  _Float16* kTf  = (_Float16*)(ws + kt_off);
  _Float16* yf   = (_Float16*)(ws + y_off);
  float*    x1   = (float*)(ws + x1_off);
  _Float16* h2f  = (_Float16*)(ws + h2_off);
  _Float16* mf   = (_Float16*)(ws + m_off);
  _Float16* wq16 = (_Float16*)(ws + wq_off);
  _Float16* wk16 = (_Float16*)(ws + wk_off);
  _Float16* wv16 = (_Float16*)(ws + wv_off);
  _Float16* wo16 = (_Float16*)(ws + wo_off);
  _Float16* wm116 = (_Float16*)(ws + wm1_off);
  _Float16* wm216 = (_Float16*)(ws + wm2_off);

  int E = in_sizes[1] / 2;  // 135168 edges (rand + self loops)

  // 1) mask bitmap
  zero_u32<<<2048, 256, 0, stream>>>(maskw, NN * NN / 32);
  build_mask<<<(E + 255) / 256, 256, 0, stream>>>(ei, E, maskw);

  // 2) weights -> f16
  cvt_f16<<<DM * DM / 256, 256, 0, stream>>>(Wq, wq16, DM * DM);
  cvt_f16<<<DM * DM / 256, 256, 0, stream>>>(Wk, wk16, DM * DM);
  cvt_f16<<<DM * DM / 256, 256, 0, stream>>>(Wv, wv16, DM * DM);
  cvt_f16<<<DM * DM / 256, 256, 0, stream>>>(Wo, wo16, DM * DM);
  cvt_f16<<<DM * 2 * DM / 256, 256, 0, stream>>>(Wm1, wm116, DM * 2 * DM);
  cvt_f16<<<DM * 2 * DM / 256, 256, 0, stream>>>(Wm2, wm216, 2 * DM * DM);

  // 3) LN1
  ln_kernel<<<NN / 8, 256, 0, stream>>>(x, g1, b1, hf);

  // 4) QKV projections (f16 out); B staged in LDS via TDM (K*Nc*2 bytes)
  size_t lds_dd = (size_t)DM * DM * 2;       // 128 KB
  size_t lds_dd2 = (size_t)DM * 2 * DM * 2;  // 256 KB
  gemm_kernel<<<128, 256, lds_dd, stream>>>(hf, wq16, bq, nullptr, qf, nullptr, DM, DM, 0);
  gemm_kernel<<<128, 256, lds_dd, stream>>>(hf, wk16, bk, nullptr, kf, nullptr, DM, DM, 0);
  gemm_kernel<<<128, 256, lds_dd, stream>>>(hf, wv16, bv, nullptr, vf, nullptr, DM, DM, 0);

  // 5) transpose K for contiguous attention B-fragments
  transpose_k<<<NN * DM / 256, 256, 0, stream>>>(kf, kTf);

  // 6) masked attention
  attn_kernel<<<dim3(NN / 16, NH), 32, 0, stream>>>(qf, kTf, vf, maskb, yf);

  // 7) output projection + residual (fp32)
  gemm_kernel<<<128, 256, lds_dd, stream>>>(yf, wo16, bo, x, nullptr, x1, DM, DM, 0);

  // 8) LN2
  ln_kernel<<<NN / 8, 256, 0, stream>>>(x1, g2, b2, h2f);

  // 9) MLP up + GELU (f16 out)
  gemm_kernel<<<256, 256, lds_dd2, stream>>>(h2f, wm116, bm1, nullptr, mf, nullptr, DM, 2 * DM, 1);

  // 10) MLP down + residual -> d_out (fp32)
  gemm_kernel<<<128, 256, lds_dd2, stream>>>(mf, wm216, bm2, x1, nullptr, (float*)d_out, 2 * DM, DM, 0);
}
